// LearnedWormholeRouter_29222957481984
// MI455X (gfx1250) — compile-verified, matrix operand order checked by reading, not run on hardware
//
#include <hip/hip_runtime.h>
#include <hip/hip_bf16.h>

typedef __attribute__((ext_vector_type(16))) _Float16 v16h;
typedef __attribute__((ext_vector_type(8)))  float    v8f;

namespace {
constexpr int   Bn       = 4;
constexpr int   Pn       = 4096;
constexpr int   Dn       = 256;
constexpr int   TK       = 32;     // top-k
constexpr int   NC       = 64;     // key columns per chunk
constexpr int   MR       = 64;     // query rows per block (16 waves)
constexpr int   NTHR     = 512;
constexpr float NEGV     = -1.0e9f;
constexpr float INV_TEMP = 10.0f;  // 1 / 0.1
}

// ---------------------------------------------------------------------------
// Kernel 1: q = l2norm(x W_q^T + b_q), k = l2norm(x W_k^T + b_k), stored f16.
// One 256-thread block per (b, p) row; thread d owns output channel d.
// ---------------------------------------------------------------------------
__global__ __launch_bounds__(256) void proj_l2_kernel(
    const float* __restrict__ x,  const float* __restrict__ Wq,
    const float* __restrict__ bq, const float* __restrict__ Wk,
    const float* __restrict__ bk,
    _Float16* __restrict__ qh, _Float16* __restrict__ kh)
{
  __shared__ float  xs[Dn];
  __shared__ float2 red[256];

  const int row = blockIdx.x;            // 0 .. B*P-1
  const int b   = row / Pn;
  const int p   = row - b * Pn;
  const int tid = threadIdx.x;

  // xp = x[:, 1:, :]  -> skip token 0
  const float* xr = x + ((size_t)b * (Pn + 1) + (size_t)p + 1) * Dn;
  xs[tid] = xr[tid];
  __syncthreads();

  float aq = bq[tid];
  float ak = bk[tid];
  const float4* wq4 = (const float4*)(Wq + (size_t)tid * Dn);
  const float4* wk4 = (const float4*)(Wk + (size_t)tid * Dn);
#pragma unroll 8
  for (int j = 0; j < Dn / 4; ++j) {
    const float4 wq = wq4[j];
    const float4 wk = wk4[j];
    const float x0 = xs[4 * j + 0], x1 = xs[4 * j + 1];
    const float x2 = xs[4 * j + 2], x3 = xs[4 * j + 3];
    aq = fmaf(x3, wq.w, fmaf(x2, wq.z, fmaf(x1, wq.y, fmaf(x0, wq.x, aq))));
    ak = fmaf(x3, wk.w, fmaf(x2, wk.z, fmaf(x1, wk.y, fmaf(x0, wk.x, ak))));
  }

  red[tid] = make_float2(aq * aq, ak * ak);
  __syncthreads();
#pragma unroll
  for (int s = 128; s > 0; s >>= 1) {
    if (tid < s) {
      red[tid].x += red[tid + s].x;
      red[tid].y += red[tid + s].y;
    }
    __syncthreads();
  }
  const float nq = fmaxf(sqrtf(red[0].x), 1e-12f);
  const float nk = fmaxf(sqrtf(red[0].y), 1e-12f);

  const size_t o = (size_t)row * Dn + tid;
  qh[o] = (_Float16)(aq / nq);
  kh[o] = (_Float16)(ak / nk);
}

// ---------------------------------------------------------------------------
// Kernel 2: scores = q k^T + bias (diag = NEG), per-row top-32 + softmax.
// 512 threads = 16 wave32s; block owns 64 query rows of one batch.
// Wave w computes the (w/4, w%4) 16x16 tile of each 64x64 score chunk with
// v_wmma_f32_16x16x32_f16. A fragment register-resident; B fragments and
// double-buffered, bias prefetched before the WMMAs, so loads overlap compute.
// ---------------------------------------------------------------------------
__global__ __launch_bounds__(NTHR) void score_topk_kernel(
    const _Float16* __restrict__ qh, const _Float16* __restrict__ kh,
    const float* __restrict__ bias,
    int* __restrict__ routes, float* __restrict__ weights)
{
  __shared__ float s_scores[MR * NC];   // 16 KB staged score chunk
  __shared__ float s_topv[MR * TK];     // 8 KB per-row top values
  __shared__ int   s_topi[MR * TK];     // 8 KB per-row top indices
  __shared__ float s_pmax[NTHR];        // 2 KB prefilter partial maxima

  const int b    = blockIdx.y;
  const int m0   = blockIdx.x * MR;     // query row base in this batch
  const int tid  = threadIdx.x;
  const int lane = tid & 31;
  const int wave = tid >> 5;            // 0..15
  const int mt   = wave >> 2;           // 0..3  (16-row sub-tile)
  const int nt   = wave & 3;            // 0..3  (16-col sub-tile)

  if (tid < MR) {
#pragma unroll
    for (int j = 0; j < TK; ++j) {
      s_topv[tid * TK + j] = -1.0e30f;
      s_topi[tid * TK + j] = 0;
    }
  }
  __syncthreads();

  // ---- A fragment: q rows [m0+mt*16, +16), all K=256, register resident.
  // A/B k order used here (contiguous): lanes 0-15 -> M=lane, K lo 16; lanes 16-31 hi 16.
  const int lrow  = lane & 15;
  const int khalf = (lane >> 4) << 4;   // 0 or 16
  const _Float16* qbase =
      qh + ((size_t)b * Pn + m0 + mt * 16 + lrow) * Dn;
  v16h a_reg[8];
#pragma unroll
  for (int kk = 0; kk < 8; ++kk)
    a_reg[kk] = *(const v16h*)(qbase + kk * 32 + khalf);

  // per-lane constant pieces of the C-tile scatter
  const int col   = nt * 16 + (lane & 15);       // column within chunk
  const int rbase = mt * 16 + ((lane >> 4) << 3); // row base within block

  const _Float16* kbatch = kh + (size_t)b * Pn * Dn;

  // ---- prologue: load B fragments for chunk 0
  v16h b_cur[8];
  {
    const _Float16* kb = kbatch + (size_t)(nt * 16 + lrow) * Dn;
#pragma unroll
    for (int kk = 0; kk < 8; ++kk)
      b_cur[kk] = *(const v16h*)(kb + kk * 32 + khalf);
  }

  float minv = -1.0e30f;   // owner-thread (tid<MR) running top-k min
  int   minpos = 0;

  for (int c = 0; c < Pn; c += NC) {
    // ---- prefetch bias for THIS chunk (consumed after the WMMAs)
    float bv[8];
#pragma unroll
    for (int r = 0; r < 8; ++r)
      bv[r] = bias[(size_t)(m0 + rbase + r) * Pn + (c + col)];

    // ---- prefetch B fragments for NEXT chunk (consumed next iteration)
    v16h b_nxt[8];
    const int cn = c + NC;
    if (cn < Pn) {
      const _Float16* kb = kbatch + (size_t)(cn + nt * 16 + lrow) * Dn;
#pragma unroll
      for (int kk = 0; kk < 8; ++kk)
        b_nxt[kk] = *(const v16h*)(kb + kk * 32 + khalf);
    } else {
#pragma unroll
      for (int kk = 0; kk < 8; ++kk) b_nxt[kk] = b_cur[kk];
    }

    // ---- 16x16 WMMA tile, K-loop over D=256 (back-to-back issues)
    v8f acc = {};
#pragma unroll
    for (int kk = 0; kk < 8; ++kk)
      acc = __builtin_amdgcn_wmma_f32_16x16x32_f16(
          false, a_reg[kk], false, b_cur[kk], (short)0, acc, false, false);

    // ---- scatter C tile to LDS with bias + diagonal mask.
    // C layout: lanes 0-15, VGPR r -> M=r; lanes 16-31, VGPR r -> M=r+8.
#pragma unroll
    for (int r = 0; r < 8; ++r) {
      const int row  = rbase + r;                // 0..63 within block
      const int grow = m0 + row;                 // global query index
      const int gcol = c + col;                  // global key index
      float s = acc[r] + bv[r];
      if (grow == gcol) s = NEGV;
      s_scores[row * NC + col] = s;
    }
    __syncthreads();

    // ---- parallel prefilter: 8 threads per row, 8 columns each
    {
      const int prow = tid >> 3;                 // 0..63
      const int psub = tid & 7;                  // 0..7
      const float* sr = s_scores + prow * NC + psub * 8;
      float pm = sr[0];
#pragma unroll
      for (int j = 1; j < 8; ++j) pm = fmaxf(pm, sr[j]);
      s_pmax[tid] = pm;
    }
    __syncthreads();

    // ---- top-k maintenance: thread t < 64 owns query row t
    if (tid < MR) {
      float rmax = s_pmax[tid * 8];
#pragma unroll
      for (int j = 1; j < 8; ++j) rmax = fmaxf(rmax, s_pmax[tid * 8 + j]);
      if (rmax > minv) {
        const float* srow = s_scores + tid * NC;
        float* tv = s_topv + tid * TK;
        int*   ti = s_topi + tid * TK;
        for (int j = 0; j < NC; ++j) {
          const float s = srow[j];
          if (s > minv) {
            tv[minpos] = s;
            ti[minpos] = c + j;
            float mv = tv[0];
            int   mp = 0;
#pragma unroll
            for (int t = 1; t < TK; ++t) {
              const float v = tv[t];
              if (v < mv) { mv = v; mp = t; }
            }
            minv = mv;
            minpos = mp;
          }
        }
      }
    }
    __syncthreads();

#pragma unroll
    for (int kk = 0; kk < 8; ++kk) b_cur[kk] = b_nxt[kk];
  }

  // ---- finalize: sort descending, softmax(score/TEMP), emit outputs
  if (tid < MR) {
    float* tv = s_topv + tid * TK;
    int*   ti = s_topi + tid * TK;
    for (int i = 1; i < TK; ++i) {
      const float v  = tv[i];
      const int   ix = ti[i];
      int j = i - 1;
      while (j >= 0 && tv[j] < v) {
        tv[j + 1] = tv[j];
        ti[j + 1] = ti[j];
        --j;
      }
      tv[j + 1] = v;
      ti[j + 1] = ix;
    }
    const float vmax = tv[0];
    float sum = 0.0f;
    for (int j = 0; j < TK; ++j)
      sum += __expf((tv[j] - vmax) * INV_TEMP);
    const float inv = 1.0f / sum;
    const size_t obase = ((size_t)b * Pn + m0 + tid) * TK;
    for (int j = 0; j < TK; ++j) {
      routes[obase + j]  = ti[j];
      weights[obase + j] = __expf((tv[j] - vmax) * INV_TEMP) * inv;
    }
  }
}

// ---------------------------------------------------------------------------
extern "C" void kernel_launch(void* const* d_in, const int* in_sizes, int n_in,
                              void* d_out, int out_size, void* d_ws, size_t ws_size,
                              hipStream_t stream) {
  (void)in_sizes; (void)n_in; (void)out_size; (void)ws_size;

  const float* x    = (const float*)d_in[0];
  const float* Wq   = (const float*)d_in[1];
  const float* bq   = (const float*)d_in[2];
  const float* Wk   = (const float*)d_in[3];
  const float* bk   = (const float*)d_in[4];
  const float* bias = (const float*)d_in[5];

  // output: routes (int32, B*P*K) then weights (f32, B*P*K)
  int*   routes  = (int*)d_out;
  float* weights = (float*)d_out + (size_t)Bn * Pn * TK;

  // workspace: f16 q and k, 8 MB each
  _Float16* qh = (_Float16*)d_ws;
  _Float16* kh = qh + (size_t)Bn * Pn * Dn;

  proj_l2_kernel<<<Bn * Pn, 256, 0, stream>>>(x, Wq, bq, Wk, bk, qh, kh);
  score_topk_kernel<<<dim3(Pn / MR, Bn), NTHR, 0, stream>>>(qh, kh, bias,
                                                            routes, weights);
}
